// LSTMBaseline_58712202936536
// MI455X (gfx1250) — compile-verified, matrix operand order checked by reading, not run on hardware
//
#include <hip/hip_runtime.h>
#include <hip/hip_bf16.h>

// ---------------------------------------------------------------------------
// Types / WMMA helpers (gfx1250, wave32)
// ---------------------------------------------------------------------------
typedef __bf16 v16bf __attribute__((ext_vector_type(16)));
typedef float  v8f   __attribute__((ext_vector_type(8)));

union FragU { v16bf v; unsigned int u[8]; };

__device__ __forceinline__ v8f wmma_bf16(v16bf a, v16bf b, v8f c) {
    // 8 args: (neg_a, A, neg_b, B, c_mod, C, reuse_a, reuse_b)
    return __builtin_amdgcn_wmma_f32_16x16x32_bf16(false, a, false, b,
                                                   (short)0, c, false, false);
}

// A/B fragment for 16-bit 16x32 tile (ISA 7.12.2):
// lanes 0-15 : row = laneM, dword d holds K pair {0,2,4,6,16,18,20,22}[d]
// lanes16-31 : row = laneM, same pattern + 8
// `row` must be 4-byte aligned (even bf16 element offset).
__device__ __forceinline__ v16bf load_frag_row(const __bf16* row, int halfK, int kb) {
    const unsigned int* ru = reinterpret_cast<const unsigned int*>(row + kb);
    FragU f;
#pragma unroll
    for (int d = 0; d < 4; ++d) f.u[d]     = ru[(halfK + 2 * d) >> 1];
#pragma unroll
    for (int d = 0; d < 4; ++d) f.u[4 + d] = ru[(16 + halfK + 2 * d) >> 1];
    return f.v;
}

__device__ __forceinline__ float sigmf(float x) {
    return 1.0f / (1.0f + __expf(-x));
}

// ---------------------------------------------------------------------------
// Problem constants
// ---------------------------------------------------------------------------
#define kB   32
#define kS   1024
#define kDIN 128
#define kH   256
#define kND  2
#define kHD  (kH * kND)   // 512

// ---------------------------------------------------------------------------
// fp32 -> bf16 conversion
// ---------------------------------------------------------------------------
__global__ void f2bf_kernel(const float* __restrict__ src, __bf16* __restrict__ dst, int n) {
    int i = blockIdx.x * blockDim.x + threadIdx.x;
    if (i < n) dst[i] = (__bf16)src[i];
}

// ---------------------------------------------------------------------------
// One bidirectional LSTM layer. gridDim.x = 2 (dir), blockDim = 512 (16 waves).
// Wave w owns hidden slice j in [w*16, w*16+16): computes i/f/g/o gate tiles
// for both 16-row batch tiles entirely in its own accumulators -> cell update
// needs no cross-wave exchange. h kept in LDS (bf16), c in VGPRs (fp32).
// Weights (bf16) streamed per step; they are L2-resident (<3MB/layer).
// ---------------------------------------------------------------------------
__global__ __launch_bounds__(512)
void lstm_layer_kernel(const __bf16* __restrict__ xin, int F,
                       const __bf16* __restrict__ wih_f, const __bf16* __restrict__ whh_f,
                       const float*  __restrict__ bias_f,
                       const __bf16* __restrict__ wih_b, const __bf16* __restrict__ whh_b,
                       const float*  __restrict__ bias_b,
                       __bf16* __restrict__ out_bf, float* __restrict__ out_f32)
{
    const int dir = blockIdx.x;                 // 0 = fwd, 1 = bwd
    const __bf16* wih  = dir ? wih_b  : wih_f;
    const __bf16* whh  = dir ? whh_b  : whh_f;
    const float*  bias = dir ? bias_b : bias_f;

    const int tid   = threadIdx.x;
    const int lane  = tid & 31;
    const int wave  = tid >> 5;                 // 0..15 -> j-tile
    const int laneM = lane & 15;                // row (A) / col (B,C)
    const int halfK = (lane >> 4) * 8;          // K half-select
    const int j0    = wave * 16;

    // LDS tiles, pitch padded (+2 bf16 = +1 dword) to break bank conflicts
    __shared__ __bf16 xlds[32 * (kHD + 2)];     // up to 32 x 512 input tile
    __shared__ __bf16 hlds[32 * (kH + 2)];      // 32 x 256 hidden state
    const int XP = F + 2;
    const int HP = kH + 2;
    const int ndw = 32 * (F >> 1);              // dwords per x tile

    for (int i = tid; i < 32 * HP; i += 512) hlds[i] = (__bf16)0.0f;

    // gate biases for this lane's column (hoisted: constant over time)
    const float bi = bias[0 * kH + j0 + laneM];
    const float bf = bias[1 * kH + j0 + laneM];
    const float bg = bias[2 * kH + j0 + laneM];
    const float bo = bias[3 * kH + j0 + laneM];

    float cst[16];
#pragma unroll
    for (int i = 0; i < 16; ++i) cst[i] = 0.0f;

    // stage first x tile
    {
        const int t0 = dir ? (kS - 1) : 0;
        const unsigned int* src = reinterpret_cast<const unsigned int*>(xin);
        unsigned int* dst = reinterpret_cast<unsigned int*>(xlds);
        for (int idx = tid; idx < ndw; idx += 512) {
            int b = idx / (F >> 1), kd = idx % (F >> 1);
            dst[b * (XP >> 1) + kd] = src[((size_t)b * kS + t0) * (F >> 1) + kd];
        }
    }
    __syncthreads();

    for (int s = 0; s < kS; ++s) {
        const int t = dir ? (kS - 1 - s) : s;

        // prefetch x rows two steps ahead (-> global_prefetch_b8, L2 warm)
        if (s + 2 < kS) {
            const int tp = dir ? (kS - 3 - s) : (s + 2);
            __builtin_prefetch((const char*)(xin + ((size_t)(tid & 31) * kS + tp) * F), 0, 1);
        }

        v8f acc[4][2];
#pragma unroll
        for (int m = 0; m < 2; ++m)
#pragma unroll
            for (int r = 0; r < 8; ++r) {
                acc[0][m][r] = bi; acc[1][m][r] = bf;
                acc[2][m][r] = bg; acc[3][m][r] = bo;
            }

        // ---- x contribution: g += x_t @ w_ih^T ----
        for (int kb = 0; kb < F; kb += 32) {
            v16bf a0 = load_frag_row(xlds + (size_t)laneM * XP,        halfK, kb);
            v16bf a1 = load_frag_row(xlds + (size_t)(16 + laneM) * XP, halfK, kb);
#pragma unroll
            for (int g = 0; g < 4; ++g) {
                const __bf16* wrow = wih + (size_t)(g * kH + j0 + laneM) * F;
                v16bf bfr = load_frag_row(wrow, halfK, kb);
                acc[g][0] = wmma_bf16(a0, bfr, acc[g][0]);
                acc[g][1] = wmma_bf16(a1, bfr, acc[g][1]);
            }
        }
        // ---- h contribution: g += h @ w_hh^T ----
        for (int kb = 0; kb < kH; kb += 32) {
            v16bf a0 = load_frag_row(hlds + (size_t)laneM * HP,        halfK, kb);
            v16bf a1 = load_frag_row(hlds + (size_t)(16 + laneM) * HP, halfK, kb);
#pragma unroll
            for (int g = 0; g < 4; ++g) {
                const __bf16* wrow = whh + (size_t)(g * kH + j0 + laneM) * kH;
                v16bf bfr = load_frag_row(wrow, halfK, kb);
                acc[g][0] = wmma_bf16(a0, bfr, acc[g][0]);
                acc[g][1] = wmma_bf16(a1, bfr, acc[g][1]);
            }
        }

        __syncthreads();   // all waves done reading xlds/hlds

        // ---- LSTM cell + write new h ----
#pragma unroll
        for (int m = 0; m < 2; ++m)
#pragma unroll
            for (int r = 0; r < 8; ++r) {
                float iv = acc[0][m][r], fv = acc[1][m][r];
                float gv = acc[2][m][r], ov = acc[3][m][r];
                float c = sigmf(fv) * cst[m * 8 + r] + sigmf(iv) * tanhf(gv);
                float h = sigmf(ov) * tanhf(c);
                cst[m * 8 + r] = c;
                int brow = m * 16 + r + ((lane >> 4) << 3);
                hlds[brow * HP + j0 + laneM] = (__bf16)h;
                size_t oidx = (((size_t)brow * kS) + t) * kHD + dir * kH + j0 + laneM;
                if (out_bf)  out_bf[oidx]  = (__bf16)h;
                if (out_f32) out_f32[oidx] = h;
            }

        // stage next x tile
        if (s + 1 < kS) {
            const int tn = dir ? (kS - 2 - s) : (s + 1);
            const unsigned int* src = reinterpret_cast<const unsigned int*>(xin);
            unsigned int* dst = reinterpret_cast<unsigned int*>(xlds);
            for (int idx = tid; idx < ndw; idx += 512) {
                int b = idx / (F >> 1), kd = idx % (F >> 1);
                dst[b * (XP >> 1) + kd] = src[((size_t)b * kS + tn) * (F >> 1) + kd];
            }
        }
        __syncthreads();   // new h / x visible
    }
}

// ---------------------------------------------------------------------------
// Causal attention context: one workgroup per batch, thread-per-feature
// streaming cumsum. context written as bf16 for the WMMA head GEMM.
// ---------------------------------------------------------------------------
__global__ __launch_bounds__(512)
void attn_ctx_kernel(const float* __restrict__ h, const float* __restrict__ attn_w,
                     const float* __restrict__ attn_b, __bf16* __restrict__ ctx)
{
    const int b   = blockIdx.x;
    const int tid = threadIdx.x;
    __shared__ float slds[kS];
    __shared__ float red[512];
    const float* hb = h + (size_t)b * kS * kHD;

    float lmax = -3.4e38f;
    for (int t = tid; t < kS; t += 512) {
        const float* row = hb + (size_t)t * kHD;
        float acc = 0.0f;
        for (int j = 0; j < kHD; ++j) acc += row[j] * attn_w[j];
        acc += attn_b[0];
        slds[t] = acc;
        lmax = fmaxf(lmax, acc);
    }
    red[tid] = lmax;
    __syncthreads();
    for (int off = 256; off > 0; off >>= 1) {
        if (tid < off) red[tid] = fmaxf(red[tid], red[tid + off]);
        __syncthreads();
    }
    const float m = red[0];

    const int j = tid;                 // 512 threads == 512 features
    float num = 0.0f, den = 0.0f;
    for (int t = 0; t < kS; ++t) {
        float e = __expf(slds[t] - m);
        den += e;
        num += e * hb[(size_t)t * kHD + j];
        ctx[((size_t)b * kS + t) * kHD + j] = (__bf16)(num / den);
    }
}

// ---------------------------------------------------------------------------
// Head GEMM: out[M=32768, N=128] = ctx[M,512] @ head_w[128,512]^T + head_b
// 8 waves / block, wave = N-tile, block = M-tile. Bias folded into C init.
// ---------------------------------------------------------------------------
__global__ __launch_bounds__(256)
void head_gemm_kernel(const __bf16* __restrict__ ctx, const __bf16* __restrict__ hw,
                      const float* __restrict__ hbias, float* __restrict__ out)
{
    const int tid   = threadIdx.x;
    const int lane  = tid & 31;
    const int nt    = tid >> 5;            // 0..7
    const int laneM = lane & 15;
    const int halfK = (lane >> 4) * 8;
    const int mt    = blockIdx.x;          // 0..2047

    const __bf16* arow = ctx + (size_t)(mt * 16 + laneM) * kHD;
    const __bf16* brow = hw  + (size_t)(nt * 16 + laneM) * kHD;
    const float  bias  = hbias[nt * 16 + laneM];

    v8f acc;
#pragma unroll
    for (int r = 0; r < 8; ++r) acc[r] = bias;

    for (int kb = 0; kb < kHD; kb += 32) {
        v16bf a  = load_frag_row(arow, halfK, kb);
        v16bf bf = load_frag_row(brow, halfK, kb);
        acc = wmma_bf16(a, bf, acc);
    }
#pragma unroll
    for (int r = 0; r < 8; ++r) {
        int orow = mt * 16 + r + ((lane >> 4) << 3);
        out[(size_t)orow * kDIN + nt * 16 + laneM] = acc[r];
    }
}

// ---------------------------------------------------------------------------
// Host driver
// ---------------------------------------------------------------------------
extern "C" void kernel_launch(void* const* d_in, const int* in_sizes, int n_in,
                              void* d_out, int out_size, void* d_ws, size_t ws_size,
                              hipStream_t stream)
{
    const float* x      = (const float*)d_in[0];
    const float* wih0f  = (const float*)d_in[1];
    const float* whh0f  = (const float*)d_in[2];
    const float* b0f    = (const float*)d_in[3];
    const float* wih0b  = (const float*)d_in[4];
    const float* whh0b  = (const float*)d_in[5];
    const float* b0b    = (const float*)d_in[6];
    const float* wih1f  = (const float*)d_in[7];
    const float* whh1f  = (const float*)d_in[8];
    const float* b1f    = (const float*)d_in[9];
    const float* wih1b  = (const float*)d_in[10];
    const float* whh1b  = (const float*)d_in[11];
    const float* b1b    = (const float*)d_in[12];
    const float* attn_w = (const float*)d_in[13];
    const float* attn_b = (const float*)d_in[14];
    const float* head_w = (const float*)d_in[15];
    const float* head_b = (const float*)d_in[16];
    float* out = (float*)d_out;
    (void)in_sizes; (void)n_in; (void)out_size; (void)ws_size;

    char* ws = (char*)d_ws;
    size_t off = 0;
    auto alloc = [&](size_t bytes) -> char* {
        char* p = ws + off;
        off += (bytes + 255) & ~(size_t)255;
        return p;
    };

    __bf16* xbf    = (__bf16*)alloc((size_t)kB * kS * kDIN * 2);
    __bf16* wih0fb = (__bf16*)alloc((size_t)4 * kH * kDIN * 2);
    __bf16* wih0bb = (__bf16*)alloc((size_t)4 * kH * kDIN * 2);
    __bf16* whh0fb = (__bf16*)alloc((size_t)4 * kH * kH * 2);
    __bf16* whh0bb = (__bf16*)alloc((size_t)4 * kH * kH * 2);
    __bf16* wih1fb = (__bf16*)alloc((size_t)4 * kH * kHD * 2);
    __bf16* wih1bb = (__bf16*)alloc((size_t)4 * kH * kHD * 2);
    __bf16* whh1fb = (__bf16*)alloc((size_t)4 * kH * kH * 2);
    __bf16* whh1bb = (__bf16*)alloc((size_t)4 * kH * kH * 2);
    __bf16* hwbf   = (__bf16*)alloc((size_t)kDIN * kHD * 2);
    __bf16* h0bf   = (__bf16*)alloc((size_t)kB * kS * kHD * 2);
    float*  h1f    = (float*) alloc((size_t)kB * kS * kHD * 4);
    __bf16* ctxbf  = (__bf16*)alloc((size_t)kB * kS * kHD * 2);

    auto cvt = [&](const float* s, __bf16* d, int n) {
        f2bf_kernel<<<(n + 255) / 256, 256, 0, stream>>>(s, d, n);
    };
    cvt(x,     xbf,    kB * kS * kDIN);
    cvt(wih0f, wih0fb, 4 * kH * kDIN);
    cvt(wih0b, wih0bb, 4 * kH * kDIN);
    cvt(whh0f, whh0fb, 4 * kH * kH);
    cvt(whh0b, whh0bb, 4 * kH * kH);
    cvt(wih1f, wih1fb, 4 * kH * kHD);
    cvt(wih1b, wih1bb, 4 * kH * kHD);
    cvt(whh1f, whh1fb, 4 * kH * kH);
    cvt(whh1b, whh1bb, 4 * kH * kH);
    cvt(head_w, hwbf,  kDIN * kHD);

    // Layer 0 (F=128): bf16 output feeds layer 1
    lstm_layer_kernel<<<2, 512, 0, stream>>>(xbf, kDIN,
        wih0fb, whh0fb, b0f, wih0bb, whh0bb, b0b, h0bf, (float*)nullptr);
    // Layer 1 (F=512): fp32 output feeds attention
    lstm_layer_kernel<<<2, 512, 0, stream>>>(h0bf, kHD,
        wih1fb, whh1fb, b1f, wih1bb, whh1bb, b1b, (__bf16*)nullptr, h1f);

    attn_ctx_kernel<<<kB, 512, 0, stream>>>(h1f, attn_w, attn_b, ctxbf);

    head_gemm_kernel<<<(kB * kS) / 16, 256, 0, stream>>>(ctxbf, hwbf, head_b, out);
}
